// LSTM_47356309406493
// MI455X (gfx1250) — compile-verified, matrix operand order checked by reading, not run on hardware
//
#include <hip/hip_runtime.h>

#define B_   128
#define T_   256
#define H_   512
#define C_   128
#define NWG  8
#define TPB  512
#define NSTEP (T_ - 1)      // 255 steps
#define NP   520            // padded LDS row stride in ushorts (bank-conflict free)
#define BH   (B_ * H_)

typedef __attribute__((ext_vector_type(16))) __bf16     v16bf;
typedef __attribute__((ext_vector_type(8)))  float      v8f;
typedef __attribute__((ext_vector_type(4)))  unsigned   u32x4;

union Frag { u32x4 u[2]; v16bf v; };

__device__ __forceinline__ unsigned short f2bf(float f) {
  unsigned u = __float_as_uint(f);
  u += 0x7FFFu + ((u >> 16) & 1u);   // round-to-nearest-even
  return (unsigned short)(u >> 16);
}

__device__ __forceinline__ float sigmoid_f(float x) {
  return 1.0f / (1.0f + __expf(-x));
}
__device__ __forceinline__ float tanh_f(float x) {
  return 1.0f - 2.0f / (__expf(2.0f * x) + 1.0f);
}
__device__ __forceinline__ v8f v8f_zero() { v8f z = {}; return z; }

// ---------------- init: zero barrier counter + h ping-pong buffers -------------
__global__ void init_kernel(unsigned* cnt, unsigned* hbuf_u32, int n_u32) {
  int i = blockIdx.x * blockDim.x + threadIdx.x;
  if (i == 0) *cnt = 0u;
  for (; i < n_u32; i += gridDim.x * blockDim.x) hbuf_u32[i] = 0u;
}

// ---------------- pack: fp32 weights -> transposed, gate-interleaved bf16 ------
__global__ void pack_kernel(const float* __restrict__ Whg, const float* __restrict__ Whi,
                            const float* __restrict__ Whf, const float* __restrict__ Who,
                            const float* __restrict__ Whp,
                            unsigned short* __restrict__ Wpack,
                            unsigned short* __restrict__ WhpT) {
  const int total1 = 4 * H_ * H_;   // 2048 packed rows x 512
  const int total2 = C_ * H_;       // W_hp transposed: 128 rows x 512
  int stride = gridDim.x * blockDim.x;
  for (int i = blockIdx.x * blockDim.x + threadIdx.x; i < total1 + total2; i += stride) {
    if (i < total1) {
      int p   = i >> 9;        // packed column-row index [0,2048)
      int k   = i & 511;
      int hcb = p >> 8;        // workgroup slice
      int r   = p & 255;
      int nh  = r >> 7;        // wave n-half
      int rr  = r & 127;
      int gate= rr >> 5;       // g,i,f,o
      int jj  = rr & 31;
      int j   = hcb * 64 + nh * 32 + jj;      // hidden column
      const float* W = (gate == 0) ? Whg : (gate == 1) ? Whi : (gate == 2) ? Whf : Who;
      Wpack[(size_t)p * H_ + k] = f2bf(W[(size_t)k * H_ + j]);   // transpose: row = out-col
    } else {
      int e = i - total1;
      int n = e >> 9;
      int k = e & 511;
      WhpT[(size_t)n * H_ + k] = f2bf(Whp[(size_t)k * C_ + n]);
    }
  }
}

// ---------------- persistent LSTM kernel ---------------------------------------
__global__ void __launch_bounds__(TPB, 1)
lstm_kernel(const int* __restrict__ x,
            const float* __restrict__ Wxg, const float* __restrict__ Wxi,
            const float* __restrict__ Wxf, const float* __restrict__ Wxo,
            const float* __restrict__ bg,  const float* __restrict__ bi,
            const float* __restrict__ bfv, const float* __restrict__ bo,
            const float* __restrict__ bp,
            const unsigned short* __restrict__ Wpack,
            const unsigned short* __restrict__ WhpT,
            unsigned short* __restrict__ hbuf,     // 2 * B_ * H_ bf16 (ping-pong)
            unsigned* __restrict__ cnt,
            float* __restrict__ out) {
  __shared__ __align__(16) unsigned short Wlds[256 * NP];   // 260 KB weight slice

  const int tid = threadIdx.x;
  const int hcb = blockIdx.x;          // which 64-wide hidden slice (x4 gates = 256 cols)

  // One-time: stage this WG's 256x512 bf16 weight slice into LDS (padded rows).
  {
    const u32x4* src = (const u32x4*)(Wpack + (size_t)hcb * 256 * H_);
    for (int idx = tid; idx < 256 * (H_ / 8); idx += TPB) {
      int r = idx >> 6;                 // 64 u32x4 per row
      int q = idx & 63;
      *(u32x4*)&Wlds[r * NP + q * 8] = src[r * 64 + q];
    }
  }
  __syncthreads();

  const int w    = tid >> 5;           // wave id [0,16)
  const int lane = tid & 31;
  const int mt   = w & 7;              // batch tile (16 rows)
  const int nh   = w >> 3;             // which 128-col half of the 256 cols
  const int lq   = lane & 15;
  const int hi   = lane >> 4;

  const int rowA = mt * 16 + lq;       // A-fragment source row (ISA: M = lane%16)

  // Hoisted per-lane constants: biases & hidden-column index for s in {0,1}.
  float bgr[2], bir[2], bfr[2], bor[2];
  int   jcol[2];
#pragma unroll
  for (int s = 0; s < 2; ++s) {
    int j   = hcb * 64 + nh * 32 + s * 16 + lq;
    jcol[s] = j;
    bgr[s] = bg[j]; bir[s] = bi[j]; bfr[s] = bfv[j]; bor[s] = bo[j];
  }
  int rowP[8];                         // D-matrix rows: M = v + 8*hi (ISA C/D layout)
#pragma unroll
  for (int v = 0; v < 8; ++v) rowP[v] = mt * 16 + v + hi * 8;

  float cst[2][8];                     // persistent cell state, register-resident
#pragma unroll
  for (int s = 0; s < 2; ++s)
#pragma unroll
    for (int v = 0; v < 8; ++v) cst[s][v] = 0.0f;

  for (int t = 0; t < NSTEP; ++t) {
    const unsigned short* hcur  = hbuf + (size_t)(t & 1) * BH;
    unsigned short*       hnext = hbuf + (size_t)((t + 1) & 1) * BH;

    v8f acc[4][2];
#pragma unroll
    for (int g = 0; g < 4; ++g)
#pragma unroll
      for (int s = 0; s < 2; ++s) acc[g][s] = v8f_zero();

    const unsigned short* arow = hcur + (size_t)rowA * H_;
#pragma unroll 2
    for (int kt = 0; kt < 16; ++kt) {
      int k0 = kt * 32;
      Frag a;
      a.u[0] = *(const u32x4*)(arow + k0 + 8 * hi);
      a.u[1] = *(const u32x4*)(arow + k0 + 16 + 8 * hi);
#pragma unroll
      for (int g = 0; g < 4; ++g) {
#pragma unroll
        for (int s = 0; s < 2; ++s) {
          int nl = nh * 128 + g * 32 + s * 16 + lq;   // LDS row = output column
          Frag b;
          b.u[0] = *(const u32x4*)&Wlds[nl * NP + k0 + 8 * hi];
          b.u[1] = *(const u32x4*)&Wlds[nl * NP + k0 + 16 + 8 * hi];
          acc[g][s] = __builtin_amdgcn_wmma_f32_16x16x32_bf16(
              false, a.v, false, b.v, (short)0, acc[g][s], false, false);
        }
      }
    }

    // Pointwise: embedding row-gather + bias + activations + cell update.
    int tok[8];
#pragma unroll
    for (int v = 0; v < 8; ++v) tok[v] = x[rowP[v] * T_ + t];

#pragma unroll
    for (int s = 0; s < 2; ++s) {
      int j = jcol[s];
#pragma unroll
      for (int v = 0; v < 8; ++v) {
        int tj = tok[v] * H_ + j;
        float zg = acc[0][s][v] + Wxg[tj] + bgr[s];
        float zi = acc[1][s][v] + Wxi[tj] + bir[s];
        float zf = acc[2][s][v] + Wxf[tj] + bfr[s];
        float zo = acc[3][s][v] + Wxo[tj] + bor[s];
        float gv = tanh_f(zg);
        float iv = sigmoid_f(zi);
        float fv = sigmoid_f(zf);
        float ov = sigmoid_f(zo);
        float cn = gv * iv + cst[s][v] * fv;
        cst[s][v] = cn;
        float hn = tanh_f(cn) * ov;
        hnext[(size_t)rowP[v] * H_ + j] = f2bf(hn);
      }
    }

    // Grid-wide barrier: monotonic counter (zeroed each launch by init_kernel).
    __syncthreads();
    if (tid == 0) {
      __threadfence();
      atomicAdd(cnt, 1u);
      unsigned target = (unsigned)(t + 1) * NWG;
      while (*(volatile unsigned*)cnt < target) { }
      __threadfence();
    }
    __syncthreads();
  }

  // Projection: out = h_final @ W_hp + b_p. WG hcb owns cols [hcb*16, hcb*16+16).
  if (w < 8) {
    const unsigned short* hf    = hbuf + (size_t)(NSTEP & 1) * BH;
    const unsigned short* arow2 = hf + (size_t)(w * 16 + lq) * H_;
    const unsigned short* brow  = WhpT + (size_t)(hcb * 16 + lq) * H_;
    v8f pacc = v8f_zero();
#pragma unroll 4
    for (int kt = 0; kt < 16; ++kt) {
      int k0 = kt * 32;
      Frag a, b;
      a.u[0] = *(const u32x4*)(arow2 + k0 + 8 * hi);
      a.u[1] = *(const u32x4*)(arow2 + k0 + 16 + 8 * hi);
      b.u[0] = *(const u32x4*)(brow + k0 + 8 * hi);
      b.u[1] = *(const u32x4*)(brow + k0 + 16 + 8 * hi);
      pacc = __builtin_amdgcn_wmma_f32_16x16x32_bf16(
          false, a.v, false, b.v, (short)0, pacc, false, false);
    }
    int col = hcb * 16 + lq;
    float bpv = bp[col];
#pragma unroll
    for (int v = 0; v < 8; ++v)
      out[(size_t)(w * 16 + v + hi * 8) * C_ + col] = pacc[v] + bpv;
  }
}

extern "C" void kernel_launch(void* const* d_in, const int* in_sizes, int n_in,
                              void* d_out, int out_size, void* d_ws, size_t ws_size,
                              hipStream_t stream) {
  const int*   x   = (const int*)  d_in[0];
  const float* Wxg = (const float*)d_in[1];
  const float* Whg = (const float*)d_in[2];
  const float* Wxi = (const float*)d_in[3];
  const float* Whi = (const float*)d_in[4];
  const float* Wxf = (const float*)d_in[5];
  const float* Whf = (const float*)d_in[6];
  const float* Wxo = (const float*)d_in[7];
  const float* Who = (const float*)d_in[8];
  const float* Whp = (const float*)d_in[9];
  const float* bg  = (const float*)d_in[10];
  const float* bi  = (const float*)d_in[11];
  const float* bf  = (const float*)d_in[12];
  const float* bo  = (const float*)d_in[13];
  const float* bp  = (const float*)d_in[14];
  float* out = (float*)d_out;

  char* ws = (char*)d_ws;
  unsigned*       cnt   = (unsigned*)ws;
  unsigned short* Wpack = (unsigned short*)(ws + 256);
  unsigned short* WhpT  = (unsigned short*)(ws + 256 + (size_t)4 * H_ * H_ * 2);
  unsigned short* hbuf  = (unsigned short*)(ws + 256 + (size_t)4 * H_ * H_ * 2 + (size_t)C_ * H_ * 2);

  hipLaunchKernelGGL(init_kernel, dim3(64), dim3(256), 0, stream,
                     cnt, (unsigned*)hbuf, BH /* 2*BH ushorts == BH uints */);
  hipLaunchKernelGGL(pack_kernel, dim3(512), dim3(256), 0, stream,
                     Whg, Whi, Whf, Who, Whp, Wpack, WhpT);
  hipLaunchKernelGGL(lstm_kernel, dim3(NWG), dim3(TPB), 0, stream,
                     x, Wxg, Wxi, Wxf, Wxo, bg, bi, bf, bo, bp,
                     Wpack, WhpT, hbuf, cnt, out);
}
